// StoSelfAttention_77146202571310
// MI455X (gfx1250) — matrix operand order repaired and checked by reading, not compiled, not run on hardware
//
#include <hip/hip_runtime.h>

typedef _Float16 f16;
typedef __attribute__((ext_vector_type(4)))  _Float16 v4h;
typedef __attribute__((ext_vector_type(8)))  _Float16 v8h;
typedef __attribute__((ext_vector_type(16))) _Float16 v16h;
typedef __attribute__((ext_vector_type(8)))  float    v8f;

#define H_SIZE 1024
#define S_LEN  1024
#define NH     16
#define HD     64
#define INV_TAU 1.0f
#define NEG_BIG (-1e30f)
#define LOG2E   1.44269504088896340736f
#define NLN2    (-0.693147180559945309f)

static __device__ __forceinline__ v16h cat16(v8h lo, v8h hi) {
  return __builtin_shufflevector(lo, hi, 0,1,2,3,4,5,6,7,8,9,10,11,12,13,14,15);
}

// A-matrix 16x32 f16 fragment (ISA 7.12.2): lane&15 = M row,
// K elements per lane: kb..kb+7 and kb+16..kb+23 with kb = 8*(lane>=16).
static __device__ __forceinline__ v16h load_a_frag(const f16* base, int ld, int lane) {
  int row = lane & 15;
  int kb  = (lane >> 4) * 8;
  const f16* p = base + row * ld + kb;
  v8h lo = *(const v8h*)(p);
  v8h hi = *(const v8h*)(p + 16);
  return cat16(lo, hi);
}

// B-matrix 32x16 f16 fragment: lane&15 = N col, K elements per lane:
// kb..kb+7 and kb+16..kb+23 with kb = 8*(lane>=16). Source is row-major [n][k].
static __device__ __forceinline__ v16h load_b_frag(const f16* base, int ld, int lane) {
  int n  = lane & 15;
  int kb = (lane >> 4) * 8;
  const f16* p = base + n * ld + kb;
  v8h lo = *(const v8h*)(p);
  v8h hi = *(const v8h*)(p + 16);
  return cat16(lo, hi);
}

static __device__ __forceinline__ v8f wmma16(v16h a, v16h b, v8f c) {
  return __builtin_amdgcn_wmma_f32_16x16x32_f16(false, a, false, b, (short)0, c, false, false);
}

// ---------------- f32 -> f16 conversion (vectorized) ----------------
__global__ void cvt_f32_to_f16(const float* __restrict__ in, f16* __restrict__ out, int n4) {
  int i = blockIdx.x * blockDim.x + threadIdx.x;
  if (i >= n4) return;
  float4 v = ((const float4*)in)[i];
  v4h h = { (f16)v.x, (f16)v.y, (f16)v.z, (f16)v.w };
  ((v4h*)out)[i] = h;
}

// ---------------- GEMM: Y = X @ W^T + bias ----------------
// MODE 0: dst16 = Q  [bh][s][d]   MODE 1: dst16 = K [bh][s][d]
// MODE 2: dst16 = Vt [bh][d][s]   MODE 3: dst32 = f32 (bias + residual)
template<int MODE>
__global__ __launch_bounds__(256)
void gemm_kernel(const f16* __restrict__ X, const f16* __restrict__ W,
                 const float* __restrict__ bias,
                 f16* __restrict__ dst16,
                 const float* __restrict__ resid, float* __restrict__ dst32)
{
  const int lane = threadIdx.x;
  const int wave = threadIdx.y;
  const int m0 = blockIdx.y * 256 + wave * 32;
  const int n0 = blockIdx.x * 64;

  v8f acc[2][4];
  #pragma unroll
  for (int mi = 0; mi < 2; ++mi)
    #pragma unroll
    for (int ni = 0; ni < 4; ++ni) { v8f z = {}; acc[mi][ni] = z; }

  for (int k0 = 0; k0 < H_SIZE; k0 += 32) {
    v16h a0 = load_a_frag(X + (size_t)(m0 +  0) * H_SIZE + k0, H_SIZE, lane);
    v16h a1 = load_a_frag(X + (size_t)(m0 + 16) * H_SIZE + k0, H_SIZE, lane);
    #pragma unroll
    for (int ni = 0; ni < 4; ++ni) {
      v16h b = load_b_frag(W + (size_t)(n0 + ni * 16) * H_SIZE + k0, H_SIZE, lane);
      acc[0][ni] = wmma16(a0, b, acc[0][ni]);
      acc[1][ni] = wmma16(a1, b, acc[1][ni]);
    }
  }

  const int half = lane >> 4, l16 = lane & 15;
  #pragma unroll
  for (int ni = 0; ni < 4; ++ni) {
    int n = n0 + ni * 16 + l16;
    float bv = bias[n];
    #pragma unroll
    for (int mi = 0; mi < 2; ++mi) {
      #pragma unroll
      for (int r = 0; r < 8; ++r) {
        int m = m0 + mi * 16 + r + half * 8;
        float val = acc[mi][ni][r] + bv;
        if (MODE == 3) {
          dst32[(size_t)m * H_SIZE + n] = val + resid[(size_t)m * H_SIZE + n];
        } else {
          int b = m >> 10, s = m & 1023;
          int h = n >> 6,  d = n & 63;
          if (MODE == 2)
            dst16[(((size_t)(b * NH + h)) * HD + d) * S_LEN + s] = (f16)val;
          else
            dst16[(((size_t)(b * NH + h)) * S_LEN + s) * HD + d] = (f16)val;
        }
      }
    }
  }
}

// ---------------- Fused flash-style gumbel-softmax attention ----------------
// grid (8, 32), block (32,8). One wave = 16 query rows of one (b,h).
// Softmax is evaluated in base-2 domain (invariant) so we can use the raw
// v_log_f32 / v_exp_f32 hardware ops without libm denorm range-reduction.
__global__ __launch_bounds__(256)
void attn_kernel(const f16* __restrict__ Q, const f16* __restrict__ K,
                 const f16* __restrict__ Vt, const float* __restrict__ gum,
                 const float* __restrict__ mask, f16* __restrict__ ctx)
{
  __shared__ f16 Pbuf[8][16 * 64];   // 16KB, wave-private 2KB slices
  const int lane = threadIdx.x, wave = threadIdx.y;
  const int half = lane >> 4, l16 = lane & 15;
  const int bh = blockIdx.y, b = bh >> 4, h = bh & 15;
  const int q0 = blockIdx.x * 128 + wave * 16;

  const f16* Qh = Q  + (size_t)bh * S_LEN * HD;
  const f16* Kh = K  + (size_t)bh * S_LEN * HD;
  const f16* Vh = Vt + (size_t)bh * HD * S_LEN;
  const float* G  = gum + (size_t)bh * S_LEN * S_LEN;
  const float* mk = mask + (size_t)b * S_LEN;
  f16* Pl = &Pbuf[wave][0];

  v16h aq0 = load_a_frag(Qh + (size_t)q0 * HD +  0, HD, lane);
  v16h aq1 = load_a_frag(Qh + (size_t)q0 * HD + 32, HD, lane);

  float mrow[8], lrow[8];   // running max / denom, base-2 domain
  #pragma unroll
  for (int r = 0; r < 8; ++r) { mrow[r] = NEG_BIG; lrow[r] = 0.0f; }
  v8f oacc[4];
  #pragma unroll
  for (int t = 0; t < 4; ++t) { v8f z = {}; oacc[t] = z; }

  for (int kt = 0; kt < S_LEN; kt += 64) {
    // ---- prefetch next key-block of the 128MB gumbel stream ----
    if (kt + 64 < S_LEN) {
      #pragma unroll
      for (int r = 0; r < 8; ++r) {
        int qr = q0 + r + half * 8;
        __builtin_prefetch(G + (size_t)qr * S_LEN + kt + 64 + l16 * 4, 0, 3);
      }
    }
    // ---- scores: Q @ K^T for 4 key tiles (d-contraction, 2 WMMA each) ----
    v8f s[4];
    #pragma unroll
    for (int t = 0; t < 4; ++t) {
      const f16* kb = Kh + (size_t)(kt + t * 16) * HD;
      v16h b0 = load_b_frag(kb +  0, HD, lane);
      v16h b1 = load_b_frag(kb + 32, HD, lane);
      v8f z = {};
      z    = wmma16(aq0, b0, z);
      s[t] = wmma16(aq1, b1, z);
    }
    // ---- gumbel noise + logits, base-2 domain:
    //      lg2 = ((s + g)/tau)*log2e, g*log2e = -log2(-ln u) ----
    float lg[4][8];
    #pragma unroll
    for (int t = 0; t < 4; ++t) {
      #pragma unroll
      for (int r = 0; r < 8; ++r) {
        int qr = q0 + r + half * 8;
        int kk = kt + t * 16 + l16;
        float u  = G[(size_t)qr * S_LEN + kk];
        float nl = __builtin_amdgcn_logf(u) * NLN2;      // -ln u  (always normal, >0)
        float gl = __builtin_amdgcn_logf(nl);            // log2(-ln u)
        lg[t][r] = (__builtin_fmaf(s[t][r], LOG2E, -gl)) * INV_TAU;
      }
    }
    // ---- online softmax stats (row = r + 8*half lives on 16 lanes) ----
    #pragma unroll
    for (int r = 0; r < 8; ++r) {
      float mx = fmaxf(fmaxf(lg[0][r], lg[1][r]), fmaxf(lg[2][r], lg[3][r]));
      #pragma unroll
      for (int off = 1; off < 16; off <<= 1)
        mx = fmaxf(mx, __shfl_xor(mx, off, 16));
      float nm = fmaxf(mrow[r], mx);
      float sc = __builtin_amdgcn_exp2f(mrow[r] - nm);
      mrow[r] = nm;
      lrow[r] *= sc;
      #pragma unroll
      for (int t = 0; t < 4; ++t) oacc[t][r] *= sc;
    }
    // ---- p = exp2(lg2 - m); denom uses UNMASKED p, numerator gets key mask ----
    float psum[8];
    #pragma unroll
    for (int r = 0; r < 8; ++r) psum[r] = 0.0f;
    #pragma unroll
    for (int t = 0; t < 4; ++t) {
      float km = mk[kt + t * 16 + l16];
      #pragma unroll
      for (int r = 0; r < 8; ++r) {
        float pe = __builtin_amdgcn_exp2f(lg[t][r] - mrow[r]);
        psum[r] += pe;
        Pl[(r + half * 8) * 64 + t * 16 + l16] = (f16)(pe * km);
      }
    }
    #pragma unroll
    for (int r = 0; r < 8; ++r) {
      float ps = psum[r];
      #pragma unroll
      for (int off = 1; off < 16; off <<= 1) ps += __shfl_xor(ps, off, 16);
      lrow[r] += ps;
    }
    // ---- P @ V (key-contraction from LDS relayout, Vt rows contiguous) ----
    #pragma unroll
    for (int ks = 0; ks < 2; ++ks) {
      v16h ap = load_a_frag(Pl + ks * 32, 64, lane);
      #pragma unroll
      for (int nt = 0; nt < 4; ++nt) {
        v16h bv = load_b_frag(Vh + (size_t)(nt * 16) * S_LEN + kt + ks * 32, S_LEN, lane);
        oacc[nt] = wmma16(ap, bv, oacc[nt]);
      }
    }
  }
  // ---- epilogue: /denominator, query mask, write context f16 ----
  #pragma unroll
  for (int r = 0; r < 8; ++r) {
    int qr = q0 + r + half * 8;
    float f = (1.0f / lrow[r]) * mk[qr];
    #pragma unroll
    for (int nt = 0; nt < 4; ++nt) {
      int d = nt * 16 + l16;
      ctx[((size_t)(b * S_LEN + qr)) * H_SIZE + h * HD + d] = (f16)(oacc[nt][r] * f);
    }
  }
}

// ---------------- LayerNorm: one wave per row ----------------
__global__ __launch_bounds__(256)
void ln_kernel(const float* __restrict__ y, const float* __restrict__ gamma,
               const float* __restrict__ beta, float* __restrict__ out)
{
  const int lane = threadIdx.x, wave = threadIdx.y;
  const int row = blockIdx.x * 8 + wave;
  const float* yr = y + (size_t)row * H_SIZE;
  float4 v[8];
  float sum = 0.0f, sq = 0.0f;
  #pragma unroll
  for (int j = 0; j < 8; ++j) {
    v[j] = ((const float4*)yr)[j * 32 + lane];
    sum += v[j].x + v[j].y + v[j].z + v[j].w;
    sq  += v[j].x * v[j].x + v[j].y * v[j].y + v[j].z * v[j].z + v[j].w * v[j].w;
  }
  #pragma unroll
  for (int off = 1; off < 32; off <<= 1) {
    sum += __shfl_xor(sum, off, 32);
    sq  += __shfl_xor(sq,  off, 32);
  }
  float mean = sum * (1.0f / 1024.0f);
  float var  = sq * (1.0f / 1024.0f) - mean * mean;
  float rstd = rsqrtf(var + 1e-5f);
  float* outr = out + (size_t)row * H_SIZE;
  #pragma unroll
  for (int j = 0; j < 8; ++j) {
    int idx = j * 32 + lane;
    float4 gg = ((const float4*)gamma)[idx];
    float4 bb = ((const float4*)beta)[idx];
    float4 o;
    o.x = (v[j].x - mean) * rstd * gg.x + bb.x;
    o.y = (v[j].y - mean) * rstd * gg.y + bb.y;
    o.z = (v[j].z - mean) * rstd * gg.z + bb.z;
    o.w = (v[j].w - mean) * rstd * gg.w + bb.w;
    ((float4*)outr)[idx] = o;
  }
}

extern "C" void kernel_launch(void* const* d_in, const int* in_sizes, int n_in,
                              void* d_out, int out_size, void* d_ws, size_t ws_size,
                              hipStream_t stream) {
  (void)in_sizes; (void)n_in; (void)out_size; (void)ws_size;
  const float* x     = (const float*)d_in[0];
  const float* mask  = (const float*)d_in[1];
  const float* gum   = (const float*)d_in[2];
  const float* Wq    = (const float*)d_in[3];
  const float* bq    = (const float*)d_in[4];
  const float* Wk    = (const float*)d_in[5];
  const float* bk    = (const float*)d_in[6];
  const float* Wv    = (const float*)d_in[7];
  const float* bv    = (const float*)d_in[8];
  const float* Wo    = (const float*)d_in[9];
  const float* bo    = (const float*)d_in[10];
  const float* gamma = (const float*)d_in[11];
  const float* beta  = (const float*)d_in[12];
  float* out = (float*)d_out;

  char* ws = (char*)d_ws;
  f16*   xh  = (f16*)(ws);                        // 4 MB  x  f16
  f16*   wqh = (f16*)(ws + ( 4u << 20));          // 2 MB
  f16*   wkh = (f16*)(ws + ( 6u << 20));          // 2 MB
  f16*   wvh = (f16*)(ws + ( 8u << 20));          // 2 MB
  f16*   woh = (f16*)(ws + (10u << 20));          // 2 MB
  f16*   qws = (f16*)(ws + (12u << 20));          // 4 MB  Q  [bh][s][d]
  f16*   kws = (f16*)(ws + (16u << 20));          // 4 MB  K  [bh][s][d]
  f16*   vtw = (f16*)(ws + (20u << 20));          // 4 MB  Vt [bh][d][s]
  f16*   ctx = (f16*)(ws + (24u << 20));          // 4 MB  attention output f16
  float* yw  = (float*)(ws + (28u << 20));        // 8 MB  pre-LN f32

  dim3 blk(32, 8);

  const int nX4 = 2048 * 1024 / 4;
  const int nW4 = 1024 * 1024 / 4;
  cvt_f32_to_f16<<<nX4 / 256, 256, 0, stream>>>(x,  xh,  nX4);
  cvt_f32_to_f16<<<nW4 / 256, 256, 0, stream>>>(Wq, wqh, nW4);
  cvt_f32_to_f16<<<nW4 / 256, 256, 0, stream>>>(Wk, wkh, nW4);
  cvt_f32_to_f16<<<nW4 / 256, 256, 0, stream>>>(Wv, wvh, nW4);
  cvt_f32_to_f16<<<nW4 / 256, 256, 0, stream>>>(Wo, woh, nW4);

  dim3 ggrid(16, 8);
  gemm_kernel<0><<<ggrid, blk, 0, stream>>>(xh, wqh, bq, qws, nullptr, nullptr);
  gemm_kernel<1><<<ggrid, blk, 0, stream>>>(xh, wkh, bk, kws, nullptr, nullptr);
  gemm_kernel<2><<<ggrid, blk, 0, stream>>>(xh, wvh, bv, vtw, nullptr, nullptr);

  dim3 agrid(8, 32);
  attn_kernel<<<agrid, blk, 0, stream>>>(qws, kws, vtw, gum, mask, ctx);

  gemm_kernel<3><<<ggrid, blk, 0, stream>>>(ctx, woh, bo, nullptr, x, yw);

  ln_kernel<<<256, blk, 0, stream>>>(yw, gamma, beta, out);
}
